// AttentionDe_lm_49520972923063
// MI455X (gfx1250) — matrix operand order repaired and checked by reading, not compiled, not run on hardware
//
#include <hip/hip_runtime.h>

#define BB     8
#define HH     32
#define WWID   32
#define NSP    1024      // H*W
#define NHEADS 8
#define DHEAD  64
#define INNER  512
#define QKSCALE 0.125f   // 64^-0.5

typedef __attribute__((ext_vector_type(16))) _Float16 v16h;
typedef __attribute__((ext_vector_type(8)))  _Float16 v8h;
typedef __attribute__((ext_vector_type(8)))  float    v8f;

#define WMMA_F16(a, b, c) \
  __builtin_amdgcn_wmma_f32_16x16x32_f16(false, (a), false, (b), (short)0, (c), false, false)

// ---------------------------------------------------------------------------
// f32 -> f16 elementwise (weight prep)
// ---------------------------------------------------------------------------
__global__ void cvt16(const float* __restrict__ s, _Float16* __restrict__ d, int n) {
  int i = blockIdx.x * 256 + threadIdx.x;
  if (i < n) d[i] = (_Float16)s[i];
}

// ---------------------------------------------------------------------------
// Depthwise 3x3, pad=1.  in: [B][C][32][32] f32, w: [C][1][3][3] f32
// out_t: f16, TRANSPOSED [b][n][C] so GEMM B-fragments are contiguous.
// ---------------------------------------------------------------------------
__global__ void dw3x3(const float* __restrict__ in, const float* __restrict__ w,
                      _Float16* __restrict__ out_t, int C) {
  int idx = blockIdx.x * 256 + threadIdx.x;
  int xx = idx & (WWID - 1);
  int yy = (idx >> 5) & (HH - 1);
  int bc = idx >> 10;            // b*C + c
  int c  = bc & (C - 1);         // C is a power of two
  int b  = bc / C;
  const float* wp = w + c * 9;
  const float* ip = in + (size_t)bc * NSP;
  float s = 0.f;
#pragma unroll
  for (int dy = 0; dy < 3; ++dy) {
    int y = yy + dy - 1;
    if ((unsigned)y < HH) {
#pragma unroll
      for (int dx = 0; dx < 3; ++dx) {
        int x = xx + dx - 1;
        if ((unsigned)x < WWID) s += ip[y * WWID + x] * wp[dy * 3 + dx];
      }
    }
  }
  int n = yy * WWID + xx;
  out_t[((size_t)b * NSP + n) * C + c] = (_Float16)s;
}

// ---------------------------------------------------------------------------
// Pointwise 1x1 conv as GEMM: C[b][m][n] = sum_k W[m][k] * act[b][n][k]
// One wave per 16x64 tile; all fragment loads are contiguous f16 vectors.
// mode 0: f32 [b][M][n]                                        (final output)
// mode 1: f16 head-split out0[b][head][n][dim]  (m = dim*8+head)    (q path)
// mode 2: m<512 -> out0 K[b][head][n][dim];  m>=512 -> out1 V transposed
//         [b][head][dim][n]                                        (kv path)
// ---------------------------------------------------------------------------
__global__ void pw_gemm(const _Float16* __restrict__ Wt, const _Float16* __restrict__ act_t,
                        float* __restrict__ outf, _Float16* __restrict__ out0,
                        _Float16* __restrict__ out1, int M, int K, int mode) {
  int nb = blockIdx.x * 64;
  int mb = blockIdx.y * 16;
  int b  = blockIdx.z;
  int lane = threadIdx.x;
  int ln = lane & 15, g = lane >> 4;

  const _Float16* Arow  = Wt + (size_t)(mb + ln) * K;
  const _Float16* Bbase = act_t + (size_t)b * NSP * K;

  v8f acc[4] = {};
  for (int kb = 0; kb < K; kb += 32) {
    // A fragment: K-map = h + 16g  -> two contiguous v8h
    v8h a0 = *(const v8h*)(Arow + kb + 16 * g);
    v8h a1 = *(const v8h*)(Arow + kb + 16 * g + 8);
    v16h af;
#pragma unroll
    for (int h = 0; h < 8; ++h) { af[h] = a0[h]; af[h + 8] = a1[h]; }
#pragma unroll
    for (int t = 0; t < 4; ++t) {
      // B fragment: K-map = h + 16g -> one contiguous v16h per lane
      int n = nb + t * 16 + ln;
      v16h bf = *(const v16h*)(Bbase + (size_t)n * K + kb + 16 * g);
      acc[t] = WMMA_F16(af, bf, acc[t]);
    }
  }

#pragma unroll
  for (int t = 0; t < 4; ++t) {
#pragma unroll
    for (int r = 0; r < 8; ++r) {
      int m = mb + r + 8 * g;
      int n = nb + t * 16 + ln;
      float vv = acc[t][r];
      if (mode == 0) {
        outf[((size_t)b * M + m) * NSP + n] = vv;
      } else if (mode == 1) {
        int head = m & 7, dim = m >> 3;
        out0[(((size_t)b * NHEADS + head) * NSP + n) * DHEAD + dim] = (_Float16)vv;
      } else {
        if (m < INNER) {
          int head = m & 7, dim = m >> 3;
          out0[(((size_t)b * NHEADS + head) * NSP + n) * DHEAD + dim] = (_Float16)vv;
        } else {
          int cch = m - INNER;
          int head = cch & 7, dim = cch >> 3;
          out1[(((size_t)b * NHEADS + head) * DHEAD + dim) * NSP + n] = (_Float16)vv;
        }
      }
    }
  }
}

// ---------------------------------------------------------------------------
// Flash attention: one wave per (b, head, 16-query tile).
// qh/kh: [B][heads][N][64] f16.  vt: [B][heads][64][N] f16 (transposed).
// obuf: [B][512][N] f32, channel = dim*8 + head.
// V tiles are staged into LDS with global_load_async_to_lds_b128, overlapped
// with the QK^T WMMAs + softmax, then consumed as ds_load_b128 B-fragments.
// ---------------------------------------------------------------------------
__global__ void attn_fa(const _Float16* __restrict__ qh, const _Float16* __restrict__ kh,
                        const _Float16* __restrict__ vt, float* __restrict__ obuf) {
  __shared__ _Float16 pt[16][32];                    // P tile, D-layout -> A-layout
  __shared__ __align__(32) _Float16 vtile[64 * 32];  // V tile: [d][32 keys]

  int ib   = blockIdx.x;
  int head = blockIdx.y;
  int b    = blockIdx.z;
  int lane = threadIdx.x, ln = lane & 15, g = lane >> 4;
  size_t bh = (size_t)(b * NHEADS + head);

  const _Float16* Q  = qh + bh * NSP * DHEAD;
  const _Float16* Km = kh + bh * NSP * DHEAD;
  const _Float16* Vt = vt + bh * DHEAD * NSP;

  // Q A-fragments for the two 32-wide k-steps of d=64 (reused for all keys)
  v16h qa0, qa1;
  {
    const _Float16* qrow = Q + (size_t)(ib * 16 + ln) * DHEAD;
    v8h q00 = *(const v8h*)(qrow + 16 * g);
    v8h q01 = *(const v8h*)(qrow + 16 * g + 8);
    v8h q10 = *(const v8h*)(qrow + 32 + 16 * g);
    v8h q11 = *(const v8h*)(qrow + 32 + 16 * g + 8);
#pragma unroll
    for (int h = 0; h < 8; ++h) {
      qa0[h] = q00[h]; qa0[h + 8] = q01[h];
      qa1[h] = q10[h]; qa1[h + 8] = q11[h];
    }
  }

  v8f o[4] = {};
  float mrow[8], lrow[8];
#pragma unroll
  for (int r = 0; r < 8; ++r) { mrow[r] = -3.0e38f; lrow[r] = 0.f; }

  unsigned vlds_base = (unsigned)(unsigned long long)(&vtile[0]);
  int vrow = lane >> 2, vchunk = lane & 3;   // 8 d-rows x 4 x 16B chunks / instr

  for (int jb = 0; jb < NSP; jb += 32) {
    // --- kick async V-tile copy global->LDS (hidden under QK^T + softmax) ---
#pragma unroll
    for (int i = 0; i < 8; ++i) {
      int d = i * 8 + vrow;
      const _Float16* gp = Vt + (size_t)d * NSP + jb + vchunk * 8;
      unsigned loff = vlds_base + (unsigned)(d * 64 + vchunk * 16);
      asm volatile("global_load_async_to_lds_b128 %0, %1, off"
                   :: "v"(loff), "v"(gp) : "memory");
    }

    // --- S = Q * K^T for 32 keys as two 16x16 tiles (contiguous v16h loads) ---
    v8f s0 = {}, s1 = {};
    {
      const _Float16* kr0 = Km + (size_t)(jb + ln) * DHEAD + 16 * g;
      const _Float16* kr1 = Km + (size_t)(jb + 16 + ln) * DHEAD + 16 * g;
      v16h k0 = *(const v16h*)kr0;
      v16h k1 = *(const v16h*)kr1;
      s0 = WMMA_F16(qa0, k0, s0);
      s1 = WMMA_F16(qa0, k1, s1);
      k0 = *(const v16h*)(kr0 + 32);
      k1 = *(const v16h*)(kr1 + 32);
      s0 = WMMA_F16(qa1, k0, s0);
      s1 = WMMA_F16(qa1, k1, s1);
    }

    // --- online softmax in C/D layout: row = r + 8g, cols across 16 lanes ---
#pragma unroll
    for (int r = 0; r < 8; ++r) {
      float v0 = s0[r] * QKSCALE;
      float v1 = s1[r] * QKSCALE;
      float mx = fmaxf(v0, v1);
#pragma unroll
      for (int msk = 1; msk < 16; msk <<= 1)
        mx = fmaxf(mx, __shfl_xor(mx, msk, 32));
      float mn    = fmaxf(mrow[r], mx);
      float alpha = __expf(mrow[r] - mn);
      mrow[r] = mn;
      float p0 = __expf(v0 - mn);
      float p1 = __expf(v1 - mn);
      float rs = p0 + p1;
#pragma unroll
      for (int msk = 1; msk < 16; msk <<= 1)
        rs += __shfl_xor(rs, msk, 32);
      lrow[r] = lrow[r] * alpha + rs;
#pragma unroll
      for (int t = 0; t < 4; ++t) o[t][r] *= alpha;
      pt[r + 8 * g][ln]      = (_Float16)p0;
      pt[r + 8 * g][16 + ln] = (_Float16)p1;
    }
    __syncthreads();

    // --- P as A fragment from LDS ---
    v16h pa;
    {
      v8h p0 = *(const v8h*)(&pt[ln][16 * g]);
      v8h p1 = *(const v8h*)(&pt[ln][16 * g + 8]);
#pragma unroll
      for (int h = 0; h < 8; ++h) { pa[h] = p0[h]; pa[h + 8] = p1[h]; }
    }

    // --- wait for async V tile, then O += P * V (4 d-tiles) ---
    asm volatile("s_wait_asynccnt 0x0" ::: "memory");
#pragma unroll
    for (int t = 0; t < 4; ++t) {
      int d = t * 16 + ln;
      v16h vb = *(const v16h*)(&vtile[d * 32 + 16 * g]);
      o[t] = WMMA_F16(pa, vb, o[t]);
    }
    __syncthreads();
  }

  // --- normalize and scatter to channel layout c = dim*8 + head ---
#pragma unroll
  for (int t = 0; t < 4; ++t) {
#pragma unroll
    for (int r = 0; r < 8; ++r) {
      int i = ib * 16 + r + 8 * g;
      int d = t * 16 + ln;
      int c = d * NHEADS + head;
      obuf[((size_t)(b * INNER + c)) * NSP + i] = o[t][r] / lrow[r];
    }
  }
}

// ---------------------------------------------------------------------------
extern "C" void kernel_launch(void* const* d_in, const int* in_sizes, int n_in,
                              void* d_out, int out_size, void* d_ws, size_t ws_size,
                              hipStream_t stream) {
  const float* q      = (const float*)d_in[0];
  const float* x      = (const float*)d_in[1];
  const float* q_dw   = (const float*)d_in[2];
  const float* q_pw   = (const float*)d_in[3];
  const float* kv_dw  = (const float*)d_in[4];
  const float* kv_pw  = (const float*)d_in[5];
  const float* out_dw = (const float*)d_in[6];
  const float* out_pw = (const float*)d_in[7];

  char* ws = (char*)d_ws;
  size_t off = 0;
  _Float16* wq16  = (_Float16*)(ws + off); off += (size_t)512  * 256 * 2;
  _Float16* wkv16 = (_Float16*)(ws + off); off += (size_t)1024 * 256 * 2;
  _Float16* wout16= (_Float16*)(ws + off); off += (size_t)256  * 512 * 2;
  _Float16* qd_t  = (_Float16*)(ws + off); off += (size_t)BB * NSP * 256   * 2; // 4 MB
  _Float16* xd_t  = (_Float16*)(ws + off); off += (size_t)BB * NSP * 256   * 2; // 4 MB
  _Float16* qhB   = (_Float16*)(ws + off); off += (size_t)BB * INNER * NSP * 2; // 8 MB
  _Float16* khB   = (_Float16*)(ws + off); off += (size_t)BB * INNER * NSP * 2; // 8 MB
  _Float16* vtB   = (_Float16*)(ws + off); off += (size_t)BB * INNER * NSP * 2; // 8 MB
  float*    obuf  = (float*)(ws + off);    off += (size_t)BB * INNER * NSP * 4; // 16 MB
  _Float16* od_t  = (_Float16*)(ws + off); off += (size_t)BB * NSP * INNER * 2; // 8 MB

  // 0) weight prep: f32 -> f16
  cvt16<<<(512 * 256) / 256, 256, 0, stream>>>(q_pw, wq16, 512 * 256);
  cvt16<<<(1024 * 256) / 256, 256, 0, stream>>>(kv_pw, wkv16, 1024 * 256);
  cvt16<<<(256 * 512) / 256, 256, 0, stream>>>(out_pw, wout16, 256 * 512);

  // 1) depthwise convs -> transposed f16 activations
  dw3x3<<<(BB * 256 * NSP) / 256, 256, 0, stream>>>(q, q_dw, qd_t, 256);
  dw3x3<<<(BB * 256 * NSP) / 256, 256, 0, stream>>>(x, kv_dw, xd_t, 256);

  // 2) pointwise GEMMs -> head-split f16 Q / K (row-major) and V (transposed)
  pw_gemm<<<dim3(NSP / 64, 512 / 16, BB), 32, 0, stream>>>(
      wq16, qd_t, nullptr, qhB, nullptr, 512, 256, 1);
  pw_gemm<<<dim3(NSP / 64, 1024 / 16, BB), 32, 0, stream>>>(
      wkv16, xd_t, nullptr, khB, vtB, 1024, 256, 2);

  // 3) flash attention (64 query tiles x 8 heads x 8 batches, one wave each)
  attn_fa<<<dim3(NSP / 16, NHEADS, BB), 32, 0, stream>>>(qhB, khB, vtB, obuf);

  // 4) output depthwise (512 ch) -> transposed f16, then final GEMM to f32
  dw3x3<<<(BB * INNER * NSP) / 256, 256, 0, stream>>>(obuf, out_dw, od_t, 512);
  pw_gemm<<<dim3(NSP / 64, 256 / 16, BB), 32, 0, stream>>>(
      wout16, od_t, (float*)d_out, nullptr, nullptr, 256, 512, 0);
}
